// GAT_90658169684149
// MI455X (gfx1250) — matrix operand order repaired and checked by reading, not compiled, not run on hardware
//
#include <hip/hip_runtime.h>

typedef __attribute__((ext_vector_type(8)))  float  v8f;
typedef __attribute__((ext_vector_type(16))) __bf16 v16bf;
typedef __attribute__((ext_vector_type(8)))  __bf16 v8bf;

static constexpr int N_NODES = 50000;
static constexpr int F_IN    = 128;
static constexpr int E_EDGES = 800000;
static constexpr int E_TOT   = E_EDGES + N_NODES;   // + self loops
static constexpr int C_HID   = 32;
static constexpr int H1      = 8;
static constexpr int HC1     = H1 * C_HID;          // 256
static constexpr float NEG_SLOPE = 0.2f;

#define DEV __device__ __forceinline__

// ---------- helpers ----------
DEV unsigned short f2bf(float f) {          // f32 -> bf16, round-to-nearest-even
  unsigned u = __float_as_uint(f);
  u += 0x7FFFu + ((u >> 16) & 1u);
  return (unsigned short)(u >> 16);
}
DEV unsigned fkey(float f) {                // order-preserving float->uint key
  unsigned u = __float_as_uint(f);
  return (u & 0x80000000u) ? ~u : (u | 0x80000000u);
}
DEV float funkey(unsigned k) {
  return __uint_as_float((k & 0x80000000u) ? (k & 0x7FFFFFFFu) : ~k);
}
DEV void edge_sd(const int* __restrict__ ei, int e, int& s, int& d) {
  if (e < E_EDGES) { s = ei[e]; d = ei[E_EDGES + e]; }
  else             { s = d = e - E_EDGES; }          // self loop
}

// ---------- utility kernels ----------
__global__ void k_fill_u32(unsigned* __restrict__ p, unsigned v, int n) {
  int i = blockIdx.x * blockDim.x + threadIdx.x;
  if (i < n) p[i] = v;
}
__global__ void k_bias_init(float* __restrict__ out, const float* __restrict__ bias,
                            int rowlen, int n) {
  int i = blockIdx.x * blockDim.x + threadIdx.x;
  if (i < n) out[i] = bias[i % rowlen];
}
__global__ void k_f32_to_bf16(const float* __restrict__ src,
                              unsigned short* __restrict__ dst, int n) {
  int i = blockIdx.x * blockDim.x + threadIdx.x;
  if (i < n) dst[i] = f2bf(src[i]);
}

// ---------- bf16 WMMA GEMM:  Y[M,OC] = Abf[M,K] @ Wbf[OC,K]^T + bias ----------
// one wave -> one 16x16 output tile; K multiple of 32; M multiple of 16.
__global__ void k_gemm_bf16_wmma(const unsigned short* __restrict__ A,
                                 const unsigned short* __restrict__ W,
                                 const float* __restrict__ bias,
                                 float* __restrict__ Y,
                                 int K, int OC) {
  const int lane = threadIdx.x & 31;
  const int wave = threadIdx.x >> 5;
  const int wpb  = blockDim.x >> 5;
  const int m0   = blockIdx.x * 16;
  const int n0   = (blockIdx.y * wpb + wave) * 16;
  const int hi   = lane >> 4;         // half-wave select (K interleave)
  const int lo   = lane & 15;

  const __bf16* Ab   = (const __bf16*)A;
  const __bf16* Wb   = (const __bf16*)W;
  const __bf16* arow = Ab + (size_t)(m0 + lo) * K;
  const __bf16* wrow = Wb + (size_t)(n0 + lo) * K;

  v8f acc = {};
  for (int k0 = 0; k0 < K; k0 += 32) {
    // A fragment: halves i -> K = k0 + 16*hi + i  (contiguous 32B)
    v16bf a = *(const v16bf*)(arow + k0 + 16 * hi);
    // B fragment: halves i -> K = k0 + 16*hi + i  (contiguous 32B)
    v16bf b = *(const v16bf*)(wrow + k0 + 16 * hi);
    acc = __builtin_amdgcn_wmma_f32_16x16x32_bf16(false, a, false, b,
                                                  (short)0, acc, false, false);
  }
  const int   col = n0 + lo;
  const float bc  = bias[col];
#pragma unroll
  for (int r = 0; r < 8; ++r)
    Y[(size_t)(m0 + r + 8 * hi) * OC + col] = acc[r] + bc;
}

// ---------- edge kernels ----------
// logit[e,h] = sum_c leaky_relu(xl[src]+xr[dst]) * att[h,c]; atomic running max per (dst,h)
__global__ void k_logit_max(const float* __restrict__ XL, const float* __restrict__ XR,
                            const float* __restrict__ att, const int* __restrict__ ei,
                            int H, float* __restrict__ logit, unsigned* __restrict__ mx,
                            int ntot) {
  int idx = blockIdx.x * blockDim.x + threadIdx.x;
  if (idx >= ntot) return;
  int e = idx / H, h = idx - e * H;
  int s, d; edge_sd(ei, e, s, d);
  const float* pl = XL + (size_t)s * H * C_HID + h * C_HID;
  const float* pr = XR + (size_t)d * H * C_HID + h * C_HID;
  const float* pa = att + h * C_HID;
  float acc = 0.f;
#pragma unroll
  for (int c = 0; c < C_HID; ++c) {
    float z = pl[c] + pr[c];
    z = (z >= 0.f) ? z : NEG_SLOPE * z;
    acc += z * pa[c];
  }
  logit[idx] = acc;
  atomicMax(&mx[(size_t)d * H + h], fkey(acc));
}

// ex = exp(logit - max[dst,h]);  logit <- ex;  denom[dst,h] += ex
__global__ void k_expsum(float* __restrict__ logit, const unsigned* __restrict__ mx,
                         float* __restrict__ den, const int* __restrict__ ei,
                         int H, int ntot) {
  int idx = blockIdx.x * blockDim.x + threadIdx.x;
  if (idx >= ntot) return;
  int e = idx / H, h = idx - e * H;
  int s, d; edge_sd(ei, e, s, d);
  float m  = funkey(mx[(size_t)d * H + h]);
  float ex = __expf(logit[idx] - m);
  logit[idx] = ex;
  atomicAdd(&den[(size_t)d * H + h], ex);
}

// out[dst,h,:] += (ex/denom) * xl[src,h,:]
__global__ void k_aggregate(const float* __restrict__ ex, const float* __restrict__ den,
                            const float* __restrict__ XL, const int* __restrict__ ei,
                            int H, float* __restrict__ out, int ntot) {
  int idx = blockIdx.x * blockDim.x + threadIdx.x;
  if (idx >= ntot) return;
  int e = idx / H, h = idx - e * H;
  int s, d; edge_sd(ei, e, s, d);
  float alpha = ex[idx] / fmaxf(den[(size_t)d * H + h], 1e-16f);
  const float* pl = XL + (size_t)s * H * C_HID + h * C_HID;
  float*       po = out + (size_t)d * H * C_HID + h * C_HID;
#pragma unroll
  for (int c = 0; c < C_HID; ++c)
    atomicAdd(&po[c], alpha * pl[c]);
}

// Hb = bf16( gelu_exact( out1 + bias1 ) )
__global__ void k_bias_gelu_bf16(const float* __restrict__ in, const float* __restrict__ bias,
                                 unsigned short* __restrict__ outb, int rowlen, int n) {
  int i = blockIdx.x * blockDim.x + threadIdx.x;
  if (i >= n) return;
  float v = in[i] + bias[i % rowlen];
  float g = 0.5f * v * (1.0f + erff(v * 0.70710678118654752f));
  outb[i] = f2bf(g);
}

// ---------- launch ----------
extern "C" void kernel_launch(void* const* d_in, const int* in_sizes, int n_in,
                              void* d_out, int out_size, void* d_ws, size_t ws_size,
                              hipStream_t stream) {
  const float* x     = (const float*)d_in[0];
  const int*   ei    = (const int*)d_in[1];
  const float* Wl1   = (const float*)d_in[2];
  const float* bl1   = (const float*)d_in[3];
  const float* Wr1   = (const float*)d_in[4];
  const float* br1   = (const float*)d_in[5];
  const float* att1  = (const float*)d_in[6];
  const float* bias1 = (const float*)d_in[7];
  const float* Wl2   = (const float*)d_in[8];
  const float* bl2   = (const float*)d_in[9];
  const float* Wr2   = (const float*)d_in[10];
  const float* br2   = (const float*)d_in[11];
  const float* att2  = (const float*)d_in[12];
  const float* bias2 = (const float*)d_in[13];
  float* out = (float*)d_out;

  char* ws = (char*)d_ws;
  size_t off = 0;
  auto take = [&](size_t bytes) -> char* {
    char* p = ws + off;
    off = (off + bytes + 255) & ~(size_t)255;
    return p;
  };
  unsigned short* Xb   = (unsigned short*)take((size_t)N_NODES * F_IN * 2);
  unsigned short* Wl1b = (unsigned short*)take((size_t)HC1 * F_IN * 2);
  unsigned short* Wr1b = (unsigned short*)take((size_t)HC1 * F_IN * 2);
  unsigned short* Wl2b = (unsigned short*)take((size_t)C_HID * HC1 * 2);
  unsigned short* Wr2b = (unsigned short*)take((size_t)C_HID * HC1 * 2);
  float*    XL1  = (float*)take((size_t)N_NODES * HC1 * 4);
  float*    XR1  = (float*)take((size_t)N_NODES * HC1 * 4);  // reused as OUT1
  float*    LOG1 = (float*)take((size_t)E_TOT * H1 * 4);
  unsigned* MAX1 = (unsigned*)take((size_t)N_NODES * H1 * 4);
  float*    DEN1 = (float*)take((size_t)N_NODES * H1 * 4);
  unsigned short* Hb = (unsigned short*)take((size_t)N_NODES * HC1 * 2);
  float*    XL2  = (float*)take((size_t)N_NODES * C_HID * 4);
  float*    XR2  = (float*)take((size_t)N_NODES * C_HID * 4);
  float*    LOG2 = (float*)take((size_t)E_TOT * 4);
  unsigned* MAX2 = (unsigned*)take((size_t)N_NODES * 4);
  float*    DEN2 = (float*)take((size_t)N_NODES * 4);
  if (off > ws_size) return;  // workspace too small: bail deterministically
  float* OUT1 = XR1;          // alias: XR1 dead after k_logit_max of layer 1

  const int TPB = 256;
  auto nb = [](int n, int tpb) { return (n + tpb - 1) / tpb; };

  // ---- convert inputs/weights to bf16 ----
  k_f32_to_bf16<<<nb(N_NODES * F_IN, TPB), TPB, 0, stream>>>(x, Xb, N_NODES * F_IN);
  k_f32_to_bf16<<<nb(HC1 * F_IN, TPB), TPB, 0, stream>>>(Wl1, Wl1b, HC1 * F_IN);
  k_f32_to_bf16<<<nb(HC1 * F_IN, TPB), TPB, 0, stream>>>(Wr1, Wr1b, HC1 * F_IN);
  k_f32_to_bf16<<<nb(C_HID * HC1, TPB), TPB, 0, stream>>>(Wl2, Wl2b, C_HID * HC1);
  k_f32_to_bf16<<<nb(C_HID * HC1, TPB), TPB, 0, stream>>>(Wr2, Wr2b, C_HID * HC1);

  // ---- layer 1 linear transforms (WMMA) ----
  {
    dim3 g(N_NODES / 16, (HC1 / 16) / 8), b(256);   // 8 waves/block, 2 col-groups
    k_gemm_bf16_wmma<<<g, b, 0, stream>>>(Xb, Wl1b, bl1, XL1, F_IN, HC1);
    k_gemm_bf16_wmma<<<g, b, 0, stream>>>(Xb, Wr1b, br1, XR1, F_IN, HC1);
  }
  // ---- layer 1 edge softmax + aggregation ----
  k_fill_u32<<<nb(N_NODES * H1, TPB), TPB, 0, stream>>>(MAX1, 0u, N_NODES * H1);
  k_fill_u32<<<nb(N_NODES * H1, TPB), TPB, 0, stream>>>((unsigned*)DEN1, 0u, N_NODES * H1);
  k_logit_max<<<nb(E_TOT * H1, TPB), TPB, 0, stream>>>(XL1, XR1, att1, ei, H1,
                                                       LOG1, MAX1, E_TOT * H1);
  k_fill_u32<<<nb(N_NODES * HC1, TPB), TPB, 0, stream>>>((unsigned*)OUT1, 0u, N_NODES * HC1);
  k_expsum<<<nb(E_TOT * H1, TPB), TPB, 0, stream>>>(LOG1, MAX1, DEN1, ei, H1, E_TOT * H1);
  k_aggregate<<<nb(E_TOT * H1, TPB), TPB, 0, stream>>>(LOG1, DEN1, XL1, ei, H1,
                                                       OUT1, E_TOT * H1);
  // ---- gelu(out1 + bias1) -> bf16 hidden ----
  k_bias_gelu_bf16<<<nb(N_NODES * HC1, TPB), TPB, 0, stream>>>(OUT1, bias1, Hb,
                                                               HC1, N_NODES * HC1);
  // ---- layer 2 linear transforms (WMMA) ----
  {
    dim3 g(N_NODES / 16, 1), b(64);                 // 2 waves cover OC=32
    k_gemm_bf16_wmma<<<g, b, 0, stream>>>(Hb, Wl2b, bl2, XL2, HC1, C_HID);
    k_gemm_bf16_wmma<<<g, b, 0, stream>>>(Hb, Wr2b, br2, XR2, HC1, C_HID);
  }
  // ---- layer 2 edge softmax + aggregation into d_out (pre-seeded with bias2) ----
  k_fill_u32<<<nb(N_NODES, TPB), TPB, 0, stream>>>(MAX2, 0u, N_NODES);
  k_fill_u32<<<nb(N_NODES, TPB), TPB, 0, stream>>>((unsigned*)DEN2, 0u, N_NODES);
  k_logit_max<<<nb(E_TOT, TPB), TPB, 0, stream>>>(XL2, XR2, att2, ei, 1,
                                                  LOG2, MAX2, E_TOT);
  k_bias_init<<<nb(N_NODES * C_HID, TPB), TPB, 0, stream>>>(out, bias2, C_HID,
                                                            N_NODES * C_HID);
  k_expsum<<<nb(E_TOT, TPB), TPB, 0, stream>>>(LOG2, MAX2, DEN2, ei, 1, E_TOT);
  k_aggregate<<<nb(E_TOT, TPB), TPB, 0, stream>>>(LOG2, DEN2, XL2, ei, 1, out, E_TOT);
}
